// AtomAttentionEncoder_10428180595083
// MI455X (gfx1250) — matrix operand order repaired and checked by reading, not compiled, not run on hardware
//
#include <hip/hip_runtime.h>
#include <hip/hip_bf16.h>
#include <math.h>
#include <stdint.h>

// ---------------------------------------------------------------------------
// AF3 atom-attention encoder for gfx1250 (MI455X).
// - All CA=128 channel contractions: v_wmma_f32_16x16x32_bf16 chains.
// - GEMM B-operand staged with CDNA5 async copies: global_load_async_to_lds_b128
//   (+ s_wait_asynccnt), from bf16 weights pre-converted once per launch.
// ---------------------------------------------------------------------------

typedef __bf16 bf16_t;
typedef __attribute__((ext_vector_type(16))) __bf16 v16bf;
typedef __attribute__((ext_vector_type(8)))  float  v8f;

#define N_ATOMS 16384
#define S_SUB   512
#define QBLK    32
#define KBLK    128
#define CA      128
#define CP      16
#define CT      384
#define NHEAD   4
#define NBLK    3
#define MKROWS  (S_SUB * KBLK)   // 65536 key rows
#define T_TOK   4096

__device__ __forceinline__ int clampi(int v, int lo, int hi) {
    return v < lo ? lo : (v > hi ? hi : v);
}
// clipped sliding window: KIDX[s,kb] = clip(s*32 - 48 + kb, 0, N-1); r = s*128+kb
__device__ __forceinline__ int kidx_of(int r) {
    int s = r >> 7, kb = r & 127;
    return clampi(s * QBLK - 48 + kb, 0, N_ATOMS - 1);
}
__device__ __forceinline__ float sigf(float x) { return 1.f / (1.f + __expf(-x)); }

__device__ __forceinline__ v8f wmma_bf16(v16bf a, v16bf b, v8f c) {
    return __builtin_amdgcn_wmma_f32_16x16x32_bf16(false, a, false, b, (short)0, c,
                                                   false, false);
}

// A-fragment K index pattern for 16-bit 16x32 A (ISA 7.12.2)
__device__ __forceinline__ int a_kpat(int e, int half) {
    int j = e >> 1, par = e & 1;
    return (j < 4) ? (8 * half + 2 * j + par) : (16 + 8 * half + 2 * (j - 4) + par);
}

// low 32 bits of a generic __shared__ pointer == LDS byte address (ISA 10.2)
__device__ __forceinline__ uint32_t lds_addr32(const void* p) {
    return (uint32_t)(uintptr_t)p;
}

// ---------------------------------------------------------------------------
// 0. f32 -> bf16 weight conversion (once per launch; makes weights DMA-able)
// ---------------------------------------------------------------------------
__global__ void cvt_bf16_kernel(const float* __restrict__ s, bf16_t* __restrict__ d,
                                size_t n)
{
    const size_t i = (size_t)blockIdx.x * blockDim.x + threadIdx.x;
    if (i < n) d[i] = (bf16_t)s[i];
}

// ---------------------------------------------------------------------------
// 1. Atom embedding
// ---------------------------------------------------------------------------
__global__ void __launch_bounds__(128)
embed_kernel(const float* __restrict__ pos, const float* __restrict__ mask,
             const int* __restrict__ element, const float* __restrict__ charge,
             const int* __restrict__ an_chars,
             const float* __restrict__ w_pos, const float* __restrict__ w_mask,
             const float* __restrict__ w_el, const float* __restrict__ w_ch,
             const float* __restrict__ w_an,
             float* __restrict__ act, float* __restrict__ qsc,
             float* __restrict__ qact)
{
    const int n = blockIdx.x, c = threadIdx.x;
    const float m  = mask[n];
    const float p0 = pos[n * 3 + 0], p1 = pos[n * 3 + 1], p2 = pos[n * 3 + 2];
    const float ch = asinhf(charge[n]);
    const int   el = element[n];
    float a = p0 * w_pos[c] + p1 * w_pos[CA + c] + p2 * w_pos[2 * CA + c]
            + m * w_mask[c] + w_el[(size_t)el * CA + c] + ch * w_ch[c];
#pragma unroll
    for (int j = 0; j < 4; ++j)
        a += w_an[(size_t)(j * 64 + an_chars[n * 4 + j]) * CA + c];
    a *= m;
    const size_t o = (size_t)n * CA + c;
    act[o] = a; qsc[o] = a; qact[o] = a;
}

// ---------------------------------------------------------------------------
// 2. Row LayerNorm (no weight)
// ---------------------------------------------------------------------------
__global__ void __launch_bounds__(128)
rownorm_kernel(const float* __restrict__ x, float* __restrict__ y)
{
    const int r = blockIdx.x, t = threadIdx.x;
    __shared__ float sd[128];
    __shared__ float smean, svar;
    const float v = x[(size_t)r * CA + t];
    sd[t] = v; __syncthreads();
    for (int o = 64; o > 0; o >>= 1) { if (t < o) sd[t] += sd[t + o]; __syncthreads(); }
    if (t == 0) smean = sd[0] * (1.f / 128.f);
    __syncthreads();
    sd[t] = v * v; __syncthreads();
    for (int o = 64; o > 0; o >>= 1) { if (t < o) sd[t] += sd[t + o]; __syncthreads(); }
    if (t == 0) svar = sd[0] * (1.f / 128.f) - smean * smean;
    __syncthreads();
    y[(size_t)r * CA + t] = (v - smean) * rsqrtf(svar + 1e-5f);
}

// ---------------------------------------------------------------------------
// 3. s2p projections: rowp = relu(act)@w_s2p_row, colp = relu(act)@w_s2p_col
// ---------------------------------------------------------------------------
__global__ void __launch_bounds__(32)
s2p_kernel(const float* __restrict__ act, const float* __restrict__ wr,
           const float* __restrict__ wc, float* __restrict__ rowp,
           float* __restrict__ colp)
{
    const int n = blockIdx.x, t = threadIdx.x;
    __shared__ float rv[128];
    for (int i = t; i < 128; i += 32) rv[i] = fmaxf(act[(size_t)n * CA + i], 0.f);
    __syncthreads();
    if (t < 16) {
        float s = 0.f;
        for (int k = 0; k < 128; ++k) s += rv[k] * wr[k * CP + t];
        rowp[n * CP + t] = s;
    } else {
        const int p = t - 16;
        float s = 0.f;
        for (int k = 0; k < 128; ++k) s += rv[k] * wc[k * CP + p];
        colp[n * CP + p] = s;
    }
}

// ---------------------------------------------------------------------------
// 4. Fused pair conditioning: features -> 3x WMMA 16x16 MLP -> residual ->
//    LN -> logits. Writes pair (d_out) and pl[NB,S,H,QB,KB] in one pass.
// ---------------------------------------------------------------------------
__global__ void __launch_bounds__(256)
pair_kernel(const float* __restrict__ pos, const int* __restrict__ uid,
            const float* __restrict__ rowp, const float* __restrict__ colp,
            const float* __restrict__ w_off, const float* __restrict__ w_dist,
            const float* __restrict__ w_mask,
            const float* __restrict__ W1, const float* __restrict__ W2,
            const float* __restrict__ W3,
            const float* __restrict__ w_ln, const float* __restrict__ w_logits,
            float* __restrict__ pair_out, float* __restrict__ pl)
{
    const int s = blockIdx.x;
    const int wave = threadIdx.x >> 5, lane = threadIdx.x & 31;
    const int half = lane >> 4, ln15 = lane & 15;
    __shared__ bf16_t W1s[256], W2s[256], W3s[256];
    __shared__ float  ptile[8][16][17];
    __shared__ float  ttile[8][16][17];
    {
        const int t = threadIdx.x;
        W1s[t] = (bf16_t)W1[t]; W2s[t] = (bf16_t)W2[t]; W3s[t] = (bf16_t)W3[t];
    }
    __syncthreads();
    // B fragments: 16x16 weights zero-padded to K=32 (rows 16..31 = 0)
    v16bf b1, b2, b3;
#pragma unroll
    for (int e = 0; e < 16; ++e) {
        const int kb = a_kpat(e, half);
        b1[e] = (kb < 16) ? W1s[kb * 16 + ln15] : (bf16_t)0.f;
        b2[e] = (kb < 16) ? W2s[kb * 16 + ln15] : (bf16_t)0.f;
        b3[e] = (kb < 16) ? W3s[kb * 16 + ln15] : (bf16_t)0.f;
    }
    for (int q = 0; q < QBLK; ++q) {
        {   // geometric + conditioning features; lane: row=ln15, chans half*8..+7
            const int m = ln15;
            const int qatom = s * QBLK + q;
            const int katom = clampi(s * QBLK - 48 + wave * 16 + m, 0, N_ATOMS - 1);
            const float o0 = pos[qatom * 3 + 0] - pos[katom * 3 + 0];
            const float o1 = pos[qatom * 3 + 1] - pos[katom * 3 + 1];
            const float o2 = pos[qatom * 3 + 2] - pos[katom * 3 + 2];
            const float sq = o0 * o0 + o1 * o1 + o2 * o2;
            const float dinv  = 1.f / (1.f + sq);
            const float valid = (uid[qatom] == uid[katom]) ? 1.f : 0.f;
#pragma unroll
            for (int cc = 0; cc < 8; ++cc) {
                const int c = half * 8 + cc;
                ptile[wave][m][c] =
                    valid * (o0 * w_off[c] + o1 * w_off[CP + c] + o2 * w_off[2 * CP + c]
                             + dinv * w_dist[c] + w_mask[c])
                    + rowp[qatom * CP + c] + colp[katom * CP + c];
            }
        }
        __syncthreads();
        // t1 = relu(pair) @ W1
        v16bf a;
#pragma unroll
        for (int e = 0; e < 16; ++e) {
            const int k = a_kpat(e, half);
            a[e] = (k < 16) ? (bf16_t)fmaxf(ptile[wave][ln15][k], 0.f) : (bf16_t)0.f;
        }
        v8f c1; for (int j = 0; j < 8; ++j) c1[j] = 0.f;
        c1 = wmma_bf16(a, b1, c1);
#pragma unroll
        for (int r = 0; r < 8; ++r) ttile[wave][r + 8 * half][ln15] = c1[r];
        __syncthreads();
        // t2 = relu(t1) @ W2
#pragma unroll
        for (int e = 0; e < 16; ++e) {
            const int k = a_kpat(e, half);
            a[e] = (k < 16) ? (bf16_t)fmaxf(ttile[wave][ln15][k], 0.f) : (bf16_t)0.f;
        }
        __syncthreads();
        v8f c2; for (int j = 0; j < 8; ++j) c2[j] = 0.f;
        c2 = wmma_bf16(a, b2, c2);
#pragma unroll
        for (int r = 0; r < 8; ++r) ttile[wave][r + 8 * half][ln15] = c2[r];
        __syncthreads();
        // t3 = relu(t2) @ W3
#pragma unroll
        for (int e = 0; e < 16; ++e) {
            const int k = a_kpat(e, half);
            a[e] = (k < 16) ? (bf16_t)fmaxf(ttile[wave][ln15][k], 0.f) : (bf16_t)0.f;
        }
        __syncthreads();
        v8f c3; for (int j = 0; j < 8; ++j) c3[j] = 0.f;
        c3 = wmma_bf16(a, b3, c3);
        // pair += t3 ; write pair out + stash for LN/logits
#pragma unroll
        for (int r = 0; r < 8; ++r) {
            const int m = r + 8 * half;
            const float fv = ptile[wave][m][ln15] + c3[r];
            ttile[wave][m][ln15] = fv;
            pair_out[((size_t)(s * QBLK + q) * KBLK + wave * 16 + m) * CP + ln15] = fv;
        }
        __syncthreads();
        // LN(w_pair_ln) + logits -> pl[i,s,h,q,k]
        if (lane < 16) {
            float mu = 0.f, s2 = 0.f;
#pragma unroll
            for (int p = 0; p < CP; ++p) { float x = ttile[wave][lane][p]; mu += x; s2 += x * x; }
            mu *= (1.f / 16.f);
            const float rstd = rsqrtf(s2 * (1.f / 16.f) - mu * mu + 1e-5f);
            float xn[CP];
#pragma unroll
            for (int p = 0; p < CP; ++p)
                xn[p] = (ttile[wave][lane][p] - mu) * rstd * w_ln[p];
#pragma unroll
            for (int j = 0; j < 12; ++j) {
                float dot = 0.f;
#pragma unroll
                for (int p = 0; p < CP; ++p) dot += xn[p] * w_logits[p * 12 + j];
                const int bi = j >> 2, hh = j & 3;
                pl[((((size_t)bi * S_SUB + s) * NHEAD + hh) * QBLK + q) * KBLK
                   + wave * 16 + lane] = dot;
            }
        }
        __syncthreads();
    }
}

// ---------------------------------------------------------------------------
// 5. Generic bf16-WMMA GEMM: Y[M,128] = epi(A[M,K] @ W[K,128] (+bias))
//    - W is bf16, DMA'd global->LDS with global_load_async_to_lds_b128
//      (ASYNCcnt), no VGPR round-trip, overlapped with A staging/convert.
//    - GATHER: A row r -> act row KIDX(r) (sliding window, inline)
//    - colscale: per-K scale folded into A staging (adaLN lnw)
//    EPI: 0=none 1=sigmoid 2=relu 3=silu
// ---------------------------------------------------------------------------
template<int EPI, bool GATHER>
__global__ void __launch_bounds__(256)
gemm128_kernel(const float* __restrict__ A, int lda,
               const float* __restrict__ colscale,
               const bf16_t* __restrict__ W, int ldw,
               const float* __restrict__ bias,
               float* __restrict__ Y, int ldy, int M, int K)
{
    __shared__ __attribute__((aligned(16))) bf16_t As[128 * 64];
    __shared__ __attribute__((aligned(16))) bf16_t Ws[64 * 128];
    const int wave = threadIdx.x >> 5;
    const int lane = threadIdx.x & 31;
    const int half = lane >> 4, ln15 = lane & 15;
    const int m0 = blockIdx.x * 128;
    const uint32_t ws_lds = lds_addr32(&Ws[0]);

    v8f acc[8];
#pragma unroll
    for (int nt = 0; nt < 8; ++nt)
#pragma unroll
        for (int j = 0; j < 8; ++j) acc[nt][j] = 0.f;

    for (int kc = 0; kc < K; kc += 64) {
        // --- async DMA of the 64x128 bf16 W chunk straight into LDS ---
        // 1024 x b128 transfers; each lane moves 16B: row = idx/16, col8 = (idx%16)*8
#pragma unroll
        for (int u = 0; u < 4; ++u) {
            const int idx = (int)threadIdx.x + u * 256;
            const int row = idx >> 4;
            const int c8  = (idx & 15) << 3;
            const uint32_t ldso = ws_lds + (uint32_t)(row * 256 + c8 * 2);
            const uint32_t goff = (uint32_t)(((kc + row) * ldw + c8) * 2);
            asm volatile("global_load_async_to_lds_b128 %0, %1, %2"
                         :: "v"(ldso), "v"(goff), "s"(W) : "memory");
        }
        if (kc + 64 < K && threadIdx.x < 128)   // prefetch next A chunk
            __builtin_prefetch(A + (size_t)(GATHER ? kidx_of(m0 + (int)threadIdx.x)
                                                   : (m0 + (int)threadIdx.x)) * lda
                                 + kc + 64, 0, 1);
        // --- A chunk: gather + lnw-scale + f32->bf16 convert (overlaps DMA) ---
        for (int t = threadIdx.x; t < 128 * 64; t += 256) {
            const int r = t >> 6, c = t & 63;
            const int row = m0 + r;
            const int src = GATHER ? kidx_of(row) : row;
            float v = A[(size_t)src * lda + kc + c];
            if (colscale) v *= colscale[kc + c];
            As[t] = (bf16_t)v;
        }
        asm volatile("s_wait_asynccnt 0x0" ::: "memory");
        __syncthreads();
#pragma unroll
        for (int ks = 0; ks < 2; ++ks) {
            v16bf af;
            {
                const int m = (wave << 4) + ln15;
#pragma unroll
                for (int e = 0; e < 16; ++e)
                    af[e] = As[m * 64 + ks * 32 + 16 * half + e];
            }
#pragma unroll
            for (int nt = 0; nt < 8; ++nt) {
                v16bf bfv;
                {
                    const int n = (nt << 4) + ln15;
#pragma unroll
                    for (int e = 0; e < 16; ++e)
                        bfv[e] = Ws[(ks * 32 + 16 * half + e) * 128 + n];
                }
                acc[nt] = wmma_bf16(af, bfv, acc[nt]);
            }
        }
        __syncthreads();
    }
#pragma unroll
    for (int nt = 0; nt < 8; ++nt)
#pragma unroll
        for (int r = 0; r < 8; ++r) {
            const int m = m0 + (wave << 4) + r + 8 * half;
            const int n = (nt << 4) + ln15;
            float v = acc[nt][r];
            if (bias) v += bias[n];
            if (EPI == 1)      v = sigf(v);
            else if (EPI == 2) v = fmaxf(v, 0.f);
            else if (EPI == 3) v = v * sigf(v);
            if (m < M) Y[(size_t)m * ldy + n] = v;
        }
}

// ---------------------------------------------------------------------------
// 6. Windowed attention: one wave per (subset, head).
// ---------------------------------------------------------------------------
__global__ void __launch_bounds__(32)
attn_kernel(const float* __restrict__ Q, const float* __restrict__ Kmat,
            const float* __restrict__ V, const float* __restrict__ PL,
            float* __restrict__ O)
{
    const int s = blockIdx.x, h = blockIdx.y;
    const int lane = threadIdx.x;
    const int half = lane >> 4, ln15 = lane & 15;
    __shared__ bf16_t qt[32 * 32];
    __shared__ bf16_t kt[128 * 32];
    __shared__ bf16_t vt[128 * 32];
    __shared__ float  lg[32 * 128];
    __shared__ bf16_t aw[32 * 128];

    for (int t = lane; t < 32 * 32; t += 32) {
        const int m = t >> 5, d = t & 31;
        qt[t] = (bf16_t)Q[(size_t)(s * QBLK + m) * CA + h * 32 + d];
    }
    for (int t = lane; t < 128 * 32; t += 32) {
        const int r = t >> 5, d = t & 31;
        kt[t] = (bf16_t)Kmat[(size_t)(s * KBLK + r) * CA + h * 32 + d];
        vt[t] = (bf16_t)V   [(size_t)(s * KBLK + r) * CA + h * 32 + d];
    }
    __syncthreads();
    const float scale = 0.17677669529663687f;  // 32^-0.5
    const float* pl = PL + (size_t)(s * NHEAD + h) * QBLK * KBLK;

#pragma unroll
    for (int mt = 0; mt < 2; ++mt) {
        v16bf af;
        {
            const int m = mt * 16 + ln15;
#pragma unroll
            for (int e = 0; e < 16; ++e) af[e] = qt[m * 32 + 16 * half + e];
        }
#pragma unroll
        for (int nt = 0; nt < 8; ++nt) {
            v16bf bfv;   // B[kd][n] = K[key=n][kd]  (transposed via indexing)
#pragma unroll
            for (int e = 0; e < 16; ++e)
                bfv[e] = kt[(nt * 16 + ln15) * 32 + 16 * half + e];
            v8f c; for (int j = 0; j < 8; ++j) c[j] = 0.f;
            c = wmma_bf16(af, bfv, c);
#pragma unroll
            for (int r = 0; r < 8; ++r) {
                const int m = mt * 16 + r + 8 * half;
                const int n = nt * 16 + ln15;
                lg[m * 128 + n] = c[r] * scale + pl[m * KBLK + n];
            }
        }
    }
    __syncthreads();
    {   // per-lane row softmax (row = lane)
        float mx = -1e30f;
        for (int k = 0; k < 128; ++k) mx = fmaxf(mx, lg[lane * 128 + k]);
        float sum = 0.f;
        for (int k = 0; k < 128; ++k) {
            const float e = __expf(lg[lane * 128 + k] - mx);
            lg[lane * 128 + k] = e; sum += e;
        }
        const float inv = 1.f / sum;
        for (int k = 0; k < 128; ++k)
            aw[lane * 128 + k] = (bf16_t)(lg[lane * 128 + k] * inv);
    }
    __syncthreads();
#pragma unroll
    for (int mt = 0; mt < 2; ++mt)
#pragma unroll
        for (int nt = 0; nt < 2; ++nt) {
            v8f c; for (int j = 0; j < 8; ++j) c[j] = 0.f;
#pragma unroll
            for (int ks = 0; ks < 4; ++ks) {
                v16bf af;
                {
                    const int m = mt * 16 + ln15;
#pragma unroll
                    for (int e = 0; e < 16; ++e)
                        af[e] = aw[m * 128 + ks * 32 + 16 * half + e];
                }
                v16bf bfv;
#pragma unroll
                for (int e = 0; e < 16; ++e)
                    bfv[e] = vt[(ks * 32 + 16 * half + e) * 32 + nt * 16 + ln15];
                c = wmma_bf16(af, bfv, c);
            }
#pragma unroll
            for (int r = 0; r < 8; ++r) {
                const int m = mt * 16 + r + 8 * half;
                O[(size_t)(s * QBLK + m) * CA + h * 32 + nt * 16 + ln15] = c[r];
            }
        }
}

// ---------------------------------------------------------------------------
// 7. Elementwise helpers
// ---------------------------------------------------------------------------
template<bool GATHER>
__global__ void combine_kernel(const float* __restrict__ g, const float* __restrict__ sv,
                               const float* __restrict__ xnorm, float* __restrict__ x,
                               int M)
{
    const size_t i = (size_t)blockIdx.x * blockDim.x + threadIdx.x;
    if (i >= (size_t)M * CA) return;
    const int r = (int)(i >> 7), c = (int)(i & 127);
    const int src = GATHER ? kidx_of(r) : r;
    x[i] = g[i] * xnorm[(size_t)src * CA + c] + sv[i];
}
__global__ void mul_kernel(float* a, const float* b, size_t n) {
    const size_t i = (size_t)blockIdx.x * blockDim.x + threadIdx.x;
    if (i < n) a[i] *= b[i];
}
__global__ void addmul_kernel(float* y, const float* a, const float* b, size_t n) {
    const size_t i = (size_t)blockIdx.x * blockDim.x + threadIdx.x;
    if (i < n) y[i] += a[i] * b[i];
}
__global__ void __launch_bounds__(384)
token_reduce_kernel(const float* __restrict__ tok, const float* __restrict__ mask,
                    float* __restrict__ out)
{
    const int t = blockIdx.x, c = threadIdx.x;
    float num = 0.f, den = 0.f;
#pragma unroll
    for (int a = 0; a < 4; ++a) {
        const float mm = mask[t * 4 + a];
        num += tok[(size_t)(t * 4 + a) * CT + c] * mm;
        den += mm;
    }
    out[(size_t)t * CT + c] = num / fmaxf(den, 1e-10f);
}

// ---------------------------------------------------------------------------
// Host orchestration
// ---------------------------------------------------------------------------
extern "C" void kernel_launch(void* const* d_in, const int* in_sizes, int n_in,
                              void* d_out, int out_size, void* d_ws, size_t ws_size,
                              hipStream_t stream)
{
    (void)in_sizes; (void)n_in; (void)out_size; (void)ws_size;
    const float* positions = (const float*)d_in[0];
    const float* maskp     = (const float*)d_in[1];
    const int*   element   = (const int*)d_in[2];
    const float* charge    = (const float*)d_in[3];
    const int*   an_chars  = (const int*)d_in[4];
    const int*   uid       = (const int*)d_in[5];
    const float* w_ref_pos       = (const float*)d_in[6];
    const float* w_ref_mask      = (const float*)d_in[7];
    const float* w_ref_element   = (const float*)d_in[8];
    const float* w_ref_charge    = (const float*)d_in[9];
    const float* w_ref_atom_name = (const float*)d_in[10];
    const float* w_s2p_row  = (const float*)d_in[11];
    const float* w_s2p_col  = (const float*)d_in[12];
    const float* w_pair_off = (const float*)d_in[13];
    const float* w_pair_dist= (const float*)d_in[14];
    const float* w_pair_mask= (const float*)d_in[15];
    const float* w_mlp1     = (const float*)d_in[16];
    const float* w_mlp2     = (const float*)d_in[17];
    const float* w_mlp3     = (const float*)d_in[18];
    const float* w_pair_ln  = (const float*)d_in[19];
    const float* w_pair_log = (const float*)d_in[20];
    const float* aq_lnw = (const float*)d_in[21];
    const float* aq_gw  = (const float*)d_in[22];
    const float* aq_gb  = (const float*)d_in[23];
    const float* aq_sw  = (const float*)d_in[24];
    const float* ak_lnw = (const float*)d_in[25];
    const float* ak_gw  = (const float*)d_in[26];
    const float* ak_gb  = (const float*)d_in[27];
    const float* ak_sw  = (const float*)d_in[28];
    const float* wq  = (const float*)d_in[29];
    const float* bq  = (const float*)d_in[30];
    const float* wk  = (const float*)d_in[31];
    const float* wv  = (const float*)d_in[32];
    const float* wg  = (const float*)d_in[33];
    const float* azi_w  = (const float*)d_in[34];
    const float* azi_cw = (const float*)d_in[35];
    const float* azi_cb = (const float*)d_in[36];
    const float* t_lnw = (const float*)d_in[37];
    const float* t_gw  = (const float*)d_in[38];
    const float* t_gb  = (const float*)d_in[39];
    const float* t_sw  = (const float*)d_in[40];
    const float* glu1  = (const float*)d_in[41];
    const float* glu2  = (const float*)d_in[42];
    const float* t_azi_w  = (const float*)d_in[43];
    const float* t_azi_cw = (const float*)d_in[44];
    const float* t_azi_cb = (const float*)d_in[45];
    const float* w_project= (const float*)d_in[46];

    // d_out layout: token_act | qact | qsc | pair
    float* outp     = (float*)d_out;
    float* tok_out  = outp;
    float* qact     = outp + (size_t)T_TOK * CT;
    float* qsc      = qact + (size_t)N_ATOMS * CA;
    float* pair_out = qsc  + (size_t)N_ATOMS * CA;

    // workspace (f32 region)
    float* wsp = (float*)d_ws;
    size_t off = 0;
    auto alloc = [&](size_t n) { float* p = wsp + off; off += n; return p; };
    const size_t ew  = (size_t)N_ATOMS * CA;   // 2M floats
    const size_t ewk = (size_t)MKROWS * CA;    // 8M floats
    float* act       = alloc(ew);
    float* act_norm  = alloc(ew);
    float* qact_norm = alloc(ew);
    float* rowp      = alloc((size_t)N_ATOMS * CP);
    float* colp      = alloc((size_t)N_ATOMS * CP);
    float* pl        = alloc((size_t)NBLK * S_SUB * NHEAD * QBLK * KBLK);
    float* xq   = alloc(ew);
    float* qbuf = alloc(ew);
    float* gq   = alloc(ew);
    float* sq   = alloc(ew);
    float* obuf = alloc(ew);
    float* o2   = alloc(ew);
    float* b1   = alloc(2 * ew);
    float* b2   = alloc(2 * ew);
    float* xk   = alloc(ewk);
    float* kb1  = alloc(ewk);   // gk -> then K
    float* kb2  = alloc(ewk);   // sk -> then V
    float* tok  = b1;           // alias: b1+b2 (8M) >= N*CT (6M); used after loop

    // bf16 weight pool (16B aligned; feeds async LDS DMA in the GEMMs)
    off = (off + 3) & ~(size_t)3;
    bf16_t* bpool = (bf16_t*)(wsp + off);
    size_t boff = 0;
    auto CW = [&](const float* src, size_t n) {
        bf16_t* p = bpool + boff;
        boff = (boff + n + 7) & ~(size_t)7;     // keep 16B alignment
        cvt_bf16_kernel<<<(n + 255) / 256, 256, 0, stream>>>(src, p, n);
        return p;
    };
    const size_t mm = (size_t)CA * CA;
    bf16_t* aq_gw_b  = CW(aq_gw,  NBLK * mm);
    bf16_t* aq_sw_b  = CW(aq_sw,  NBLK * mm);
    bf16_t* ak_gw_b  = CW(ak_gw,  NBLK * mm);
    bf16_t* ak_sw_b  = CW(ak_sw,  NBLK * mm);
    bf16_t* wq_b     = CW(wq,     NBLK * mm);
    bf16_t* wk_b     = CW(wk,     NBLK * mm);
    bf16_t* wv_b     = CW(wv,     NBLK * mm);
    bf16_t* wg_b     = CW(wg,     NBLK * mm);
    bf16_t* azi_w_b  = CW(azi_w,  NBLK * mm);
    bf16_t* azi_cw_b = CW(azi_cw, NBLK * mm);
    bf16_t* t_gw_b   = CW(t_gw,   NBLK * mm);
    bf16_t* t_sw_b   = CW(t_sw,   NBLK * mm);
    bf16_t* glu1_b   = CW(glu1,   NBLK * 2 * mm);
    bf16_t* glu2_b   = CW(glu2,   NBLK * 2 * mm);
    bf16_t* t_azi_w_b  = CW(t_azi_w,  NBLK * 2 * mm);
    bf16_t* t_azi_cw_b = CW(t_azi_cw, NBLK * mm);
    bf16_t* w_project_b = CW(w_project, (size_t)CA * CT);

    auto G = [&](int epi, bool gather, const float* A, int lda, const float* cs,
                 const bf16_t* Wm, int ldw, const float* bias, float* Y, int ldy,
                 int M, int K) {
        dim3 grid(M / 128), blk(256);
        if (!gather) {
            if (epi == 0)      gemm128_kernel<0,false><<<grid,blk,0,stream>>>(A,lda,cs,Wm,ldw,bias,Y,ldy,M,K);
            else if (epi == 1) gemm128_kernel<1,false><<<grid,blk,0,stream>>>(A,lda,cs,Wm,ldw,bias,Y,ldy,M,K);
            else if (epi == 2) gemm128_kernel<2,false><<<grid,blk,0,stream>>>(A,lda,cs,Wm,ldw,bias,Y,ldy,M,K);
            else               gemm128_kernel<3,false><<<grid,blk,0,stream>>>(A,lda,cs,Wm,ldw,bias,Y,ldy,M,K);
        } else {
            if (epi == 1)      gemm128_kernel<1,true ><<<grid,blk,0,stream>>>(A,lda,cs,Wm,ldw,bias,Y,ldy,M,K);
            else               gemm128_kernel<0,true ><<<grid,blk,0,stream>>>(A,lda,cs,Wm,ldw,bias,Y,ldy,M,K);
        }
    };
    const int EB = 256;
    const size_t gq_ew  = (ew  + EB - 1) / EB;
    const size_t gq_ewk = (ewk + EB - 1) / EB;

    // ---- stage 1: embedding, norms, pair conditioning ----
    embed_kernel<<<N_ATOMS, 128, 0, stream>>>(positions, maskp, element, charge, an_chars,
        w_ref_pos, w_ref_mask, w_ref_element, w_ref_charge, w_ref_atom_name,
        act, qsc, qact);
    rownorm_kernel<<<N_ATOMS, 128, 0, stream>>>(act, act_norm);
    s2p_kernel<<<N_ATOMS, 32, 0, stream>>>(act, w_s2p_row, w_s2p_col, rowp, colp);
    pair_kernel<<<S_SUB, 256, 0, stream>>>(positions, uid, rowp, colp,
        w_pair_off, w_pair_dist, w_pair_mask, w_mlp1, w_mlp2, w_mlp3,
        w_pair_ln, w_pair_log, pair_out, pl);

    // ---- stage 2: NB transformer blocks ----
    for (int i = 0; i < NBLK; ++i) {
        const float *aq_lnw_i = aq_lnw + (size_t)i*CA, *aq_gb_i = aq_gb + (size_t)i*CA;
        const float *ak_lnw_i = ak_lnw + (size_t)i*CA, *ak_gb_i = ak_gb + (size_t)i*CA;
        const float *bq_i = bq + (size_t)i*CA, *azi_cb_i = azi_cb + (size_t)i*CA;
        const float *t_lnw_i = t_lnw + (size_t)i*CA, *t_gb_i = t_gb + (size_t)i*CA;
        const float *t_azi_cb_i = t_azi_cb + (size_t)i*CA;
        const bf16_t *aq_gw_i = aq_gw_b + i*mm, *aq_sw_i = aq_sw_b + i*mm;
        const bf16_t *ak_gw_i = ak_gw_b + i*mm, *ak_sw_i = ak_sw_b + i*mm;
        const bf16_t *wq_i = wq_b + i*mm, *wk_i = wk_b + i*mm, *wv_i = wv_b + i*mm,
                     *wg_i = wg_b + i*mm;
        const bf16_t *azi_w_i = azi_w_b + i*mm, *azi_cw_i = azi_cw_b + i*mm;
        const bf16_t *t_gw_i = t_gw_b + i*mm, *t_sw_i = t_sw_b + i*mm;
        const bf16_t *glu1_i = glu1_b + (size_t)i*2*mm, *glu2_i = glu2_b + (size_t)i*2*mm;
        const bf16_t *t_azi_w_i = t_azi_w_b + (size_t)i*2*mm,
                     *t_azi_cw_i = t_azi_cw_b + i*mm;

        // adaLN (queries): xq = sig(c@gw+gb)*ln(qact) + c@sw, c = ln(qsc)*lnw
        rownorm_kernel<<<N_ATOMS, 128, 0, stream>>>(qact, qact_norm);
        G(1,false, act_norm,128, aq_lnw_i, aq_gw_i,128, aq_gb_i, gq,128, N_ATOMS,128);
        G(0,false, act_norm,128, aq_lnw_i, aq_sw_i,128, nullptr, sq,128, N_ATOMS,128);
        combine_kernel<false><<<gq_ew, EB, 0, stream>>>(gq, sq, qact_norm, xq, N_ATOMS);
        // adaLN (keys): rows gathered through the sliding window inline
        G(1,true,  act_norm,128, ak_lnw_i, ak_gw_i,128, ak_gb_i, kb1,128, MKROWS,128);
        G(0,true,  act_norm,128, ak_lnw_i, ak_sw_i,128, nullptr, kb2,128, MKROWS,128);
        combine_kernel<true><<<gq_ewk, EB, 0, stream>>>(kb1, kb2, qact_norm, xk, MKROWS);
        // projections
        G(0,false, xq,128, nullptr, wq_i,128, bq_i,    qbuf,128, N_ATOMS,128);
        G(0,false, xk,128, nullptr, wk_i,128, nullptr, kb1, 128, MKROWS,128);
        G(0,false, xk,128, nullptr, wv_i,128, nullptr, kb2, 128, MKROWS,128);
        G(1,false, xq,128, nullptr, wg_i,128, nullptr, gq,  128, N_ATOMS,128);
        // attention with precomputed pair logits
        attn_kernel<<<dim3(S_SUB, NHEAD), 32, 0, stream>>>(qbuf, kb1, kb2,
            pl + (size_t)i * S_SUB * NHEAD * QBLK * KBLK, obuf);
        mul_kernel<<<gq_ew, EB, 0, stream>>>(obuf, gq, ew);
        // AdaptiveZeroInit + residual
        G(0,false, obuf,128, nullptr, azi_w_i, 128, nullptr,  o2,128, N_ATOMS,128);
        G(1,false, act, 128, nullptr, azi_cw_i,128, azi_cb_i, gq,128, N_ATOMS,128);
        addmul_kernel<<<gq_ew, EB, 0, stream>>>(qact, o2, gq, ew);
        // transition: adaLN -> SwiGLU -> AZI -> residual
        rownorm_kernel<<<N_ATOMS, 128, 0, stream>>>(qact, qact_norm);
        G(1,false, act_norm,128, t_lnw_i, t_gw_i,128, t_gb_i, gq,128, N_ATOMS,128);
        G(0,false, act_norm,128, t_lnw_i, t_sw_i,128, nullptr, sq,128, N_ATOMS,128);
        combine_kernel<false><<<gq_ew, EB, 0, stream>>>(gq, sq, qact_norm, xq, N_ATOMS);
        G(3,false, xq,128, nullptr, glu1_i,      256, nullptr, b1,      256, N_ATOMS,128);
        G(3,false, xq,128, nullptr, glu1_i + 128,256, nullptr, b1 + 128,256, N_ATOMS,128);
        G(0,false, xq,128, nullptr, glu2_i,      256, nullptr, b2,      256, N_ATOMS,128);
        G(0,false, xq,128, nullptr, glu2_i + 128,256, nullptr, b2 + 128,256, N_ATOMS,128);
        mul_kernel<<<(2 * ew + EB - 1) / EB, EB, 0, stream>>>(b1, b2, 2 * ew);
        G(0,false, b1,256, nullptr, t_azi_w_i, 128, nullptr,    o2,128, N_ATOMS,256);
        G(1,false, act,128, nullptr, t_azi_cw_i,128, t_azi_cb_i, gq,128, N_ATOMS,128);
        addmul_kernel<<<gq_ew, EB, 0, stream>>>(qact, o2, gq, ew);
    }

    // ---- stage 3: projection + masked token mean ----
    G(2,false, qact,128, nullptr, w_project_b,       CT, nullptr, tok,       CT, N_ATOMS,128);
    G(2,false, qact,128, nullptr, w_project_b + 128, CT, nullptr, tok + 128, CT, N_ATOMS,128);
    G(2,false, qact,128, nullptr, w_project_b + 256, CT, nullptr, tok + 256, CT, N_ATOMS,128);
    token_reduce_kernel<<<T_TOK, 384, 0, stream>>>(tok, maskp, tok_out);
}
